// MultiHeadAttention_35545149342060
// MI455X (gfx1250) — matrix operand order repaired and checked by reading, not compiled, not run on hardware
//
#include <hip/hip_runtime.h>

// ---------------------------------------------------------------------------
// MultiHeadAttention for MI455X (gfx1250, wave32, WMMA bf16 16x16x32)
//   B=4, S=2048, Dmodel=1024, H=16, HeadDim=64
// Pipeline:
//   convert x -> bf16; convert W* -> bf16 transposed [n][k]
//   gemm_bf16 (async global->LDS double-buffered) x3 -> Q,K,V^T
//   flash attention (causal) -> ctx
//   gemm_bf16 -> out f32 + bias
// ---------------------------------------------------------------------------

#define BATCH  4
#define SLEN   2048
#define DM     1024
#define NH     16
#define HD     64
#define BSROWS (BATCH * SLEN)   // 8192

typedef __bf16 bf16;
typedef __attribute__((ext_vector_type(16))) __bf16 v16bf;
typedef __attribute__((ext_vector_type(8)))  __bf16 v8bf;
typedef __attribute__((ext_vector_type(8)))  float  v8f;

__device__ __forceinline__ v16bf concat8(v8bf lo, v8bf hi) {
  return __builtin_shufflevector(lo, hi, 0,1,2,3,4,5,6,7,8,9,10,11,12,13,14,15);
}

// LDS flat address low 32 bits == byte offset within wave's LDS allocation
__device__ __forceinline__ uint32_t to_lds(const void* p) {
  return (uint32_t)(uintptr_t)p;
}

// CDNA5 async copy: 16 bytes/lane global -> LDS, tracked by ASYNCcnt
__device__ __forceinline__ void async_copy_b128(uint32_t lds_off, const void* g) {
  asm volatile("global_load_async_to_lds_b128 %0, %1, off"
               :: "v"(lds_off), "v"(g) : "memory");
}

// ---------------------------------------------------------------------------
// f32 -> bf16 bulk convert (x): 8 elements/thread
// ---------------------------------------------------------------------------
__global__ __launch_bounds__(256) void convert_x_kernel(
    const float* __restrict__ X, bf16* __restrict__ Xb)
{
  const size_t i = ((size_t)blockIdx.x * 256 + threadIdx.x) * 8;
  const float4 a = *(const float4*)&X[i];
  const float4 b = *(const float4*)&X[i + 4];
  v8bf o;
  o[0] = (bf16)a.x; o[1] = (bf16)a.y; o[2] = (bf16)a.z; o[3] = (bf16)a.w;
  o[4] = (bf16)b.x; o[5] = (bf16)b.y; o[6] = (bf16)b.z; o[7] = (bf16)b.w;
  *(v8bf*)&Xb[i] = o;
}

// ---------------------------------------------------------------------------
// W[k][n] f32 -> Wt[n][k] bf16, 32x32 tiles through LDS (coalesced both sides)
// ---------------------------------------------------------------------------
__global__ __launch_bounds__(256) void convert_wT_kernel(
    const float* __restrict__ W, bf16* __restrict__ Wt)
{
  __shared__ bf16 t[32][33];
  const int tx = threadIdx.x;          // 0..31
  const int ty = threadIdx.y;          // 0..7
  const int n0 = blockIdx.x * 32;
  const int k0 = blockIdx.y * 32;
  #pragma unroll
  for (int i = 0; i < 4; ++i)
    t[ty + 8 * i][tx] = (bf16)W[(size_t)(k0 + ty + 8 * i) * DM + n0 + tx];
  __syncthreads();
  #pragma unroll
  for (int i = 0; i < 4; ++i)
    Wt[(size_t)(n0 + ty + 8 * i) * DM + k0 + tx] = t[tx][ty + 8 * i];
}

// ---------------------------------------------------------------------------
// Unified bf16 GEMM: C[128x128 tile] = A[M][1024] @ Wt[n][k]^T
//   mode 0: bf16 out [((b*NH+h)*SLEN+s)*HD + d]      (Q, K)
//   mode 1: bf16 out [((b*NH+h)*HD+d)*SLEN + s]      (V transposed)
//   mode 2: f32  out [m*DM + n] + bias               (output projection)
// K-step 64, double-buffered LDS filled by async global->LDS copies.
// 8 waves: wave tile 32(M) x 64(N) -> 2x4 WMMA accumulators, 16 WMMA/stage.
// ---------------------------------------------------------------------------
__global__ __launch_bounds__(256) void gemm_bf16_kernel(
    const bf16* __restrict__ A, const bf16* __restrict__ Wt,
    bf16* __restrict__ outb, float* __restrict__ outf,
    const float* __restrict__ bias, int mode)
{
  __shared__ bf16 sA[2][128][72];   // 18 KB x2
  __shared__ bf16 sB[2][128][72];   // 18 KB x2
  const int tid  = threadIdx.x;
  const int lane = tid & 31;
  const int wave = tid >> 5;
  const int wm   = wave & 3;            // M offset 32*wm
  const int wn   = wave >> 2;           // N offset 64*wn
  const int m0   = blockIdx.x * 128;
  const int n0   = blockIdx.y * 128;
  const int l15  = lane & 15;
  const int hh   = (lane >> 4) & 1;

  v8f acc[2][4] = {};

  // issue one stage: 128x64-half tiles of A and B = 8 async b128 per thread
  auto issue = [&](int buf, int k0) {
    #pragma unroll
    for (int i = 0; i < 4; ++i) {
      const int cid = tid + 256 * i;    // 0..1023
      const int row = cid >> 3;
      const int ch  = (cid & 7) * 8;    // half offset within 64-wide k slab
      async_copy_b128(to_lds(&sA[buf][row][ch]),
                      &A [(size_t)(m0 + row) * DM + k0 + ch]);
      async_copy_b128(to_lds(&sB[buf][row][ch]),
                      &Wt[(size_t)(n0 + row) * DM + k0 + ch]);
    }
  };

  issue(0, 0);
  for (int ks = 0; ks < DM / 64; ++ks) {
    const int buf = ks & 1;
    if (ks + 1 < DM / 64) {
      issue(buf ^ 1, (ks + 1) * 64);
      asm volatile("s_wait_asynccnt 0x8" ::: "memory");  // prev stage resident
    } else {
      asm volatile("s_wait_asynccnt 0x0" ::: "memory");
    }
    __syncthreads();

    #pragma unroll
    for (int c = 0; c < 2; ++c) {       // two 32-wide K chunks
      v16bf af[2], bfr[4];
      #pragma unroll
      for (int mt = 0; mt < 2; ++mt) {
        const bf16* p = &sA[buf][wm * 32 + mt * 16 + l15][c * 32 + hh * 8];
        af[mt] = concat8(*(const v8bf*)p, *(const v8bf*)(p + 16));
      }
      #pragma unroll
      for (int nt = 0; nt < 4; ++nt) {
        const bf16* p = &sB[buf][wn * 64 + nt * 16 + l15][c * 32 + hh * 8];
        bfr[nt] = concat8(*(const v8bf*)p, *(const v8bf*)(p + 16));
      }
      #pragma unroll
      for (int mt = 0; mt < 2; ++mt)
        #pragma unroll
        for (int nt = 0; nt < 4; ++nt)
          acc[mt][nt] = __builtin_amdgcn_wmma_f32_16x16x32_bf16(
              false, af[mt], false, bfr[nt], (short)0, acc[mt][nt], false, false);
    }
    __syncthreads();
  }

  #pragma unroll
  for (int mt = 0; mt < 2; ++mt)
    #pragma unroll
    for (int nt = 0; nt < 4; ++nt)
      #pragma unroll
      for (int r = 0; r < 8; ++r) {
        const int m = m0 + wm * 32 + mt * 16 + hh * 8 + r;
        const int n = n0 + wn * 64 + nt * 16 + l15;
        if (mode == 2) {
          outf[(size_t)m * DM + n] = acc[mt][nt][r] + bias[n];
        } else {
          const int b = m >> 11, s = m & (SLEN - 1);
          const int h = n >> 6,  d = n & (HD - 1);
          const int bh = b * NH + h;
          const bf16 val = (bf16)acc[mt][nt][r];
          if (mode == 0) outb[((size_t)bh * SLEN + s) * HD + d] = val;
          else           outb[((size_t)bh * HD + d) * SLEN + s] = val;
        }
      }
}

// ---------------------------------------------------------------------------
// Flash attention, causal. Wave = 16 query rows; key blocks of 32.
// Q,K: [bh][s][64] bf16; VT: [bh][64][s] bf16; CTX out: [b*S+s][1024] bf16.
// ---------------------------------------------------------------------------
__global__ __launch_bounds__(256) void attn_kernel(
    const bf16* __restrict__ Q, const bf16* __restrict__ K,
    const bf16* __restrict__ VT, bf16* __restrict__ CTX)
{
  __shared__ bf16 sP[8][16][40];   // per-wave P transpose staging
  const int tid  = threadIdx.x;
  const int lane = tid & 31;
  const int wave = tid >> 5;
  const int l15  = lane & 15;
  const int hh   = (lane >> 4) & 1;
  const int bh   = blockIdx.y;
  const int q0   = blockIdx.x * 128 + wave * 16;
  const size_t qkbase = (size_t)bh * SLEN * HD;
  const size_t vbase  = (size_t)bh * HD * SLEN;

  v16bf qf[2];
  #pragma unroll
  for (int c = 0; c < 2; ++c) {
    const bf16* p = &Q[qkbase + (size_t)(q0 + l15) * HD + c * 32 + hh * 8];
    qf[c] = concat8(*(const v8bf*)p, *(const v8bf*)(p + 16));
  }

  v8f o[4] = {};
  float mrow[8], lrow[8];
  #pragma unroll
  for (int r = 0; r < 8; ++r) { mrow[r] = -__builtin_inff(); lrow[r] = 0.f; }

  const int kend = q0 + 16;     // exclusive causal bound for this wave
  for (int kb = 0; kb < kend; kb += 32) {
    v8f sc[2] = {};
    #pragma unroll
    for (int ns = 0; ns < 2; ++ns)
      #pragma unroll
      for (int c = 0; c < 2; ++c) {
        const bf16* p = &K[qkbase + (size_t)(kb + ns * 16 + l15) * HD + c * 32 + hh * 8];
        const v16bf kf = concat8(*(const v8bf*)p, *(const v8bf*)(p + 16));
        sc[ns] = __builtin_amdgcn_wmma_f32_16x16x32_bf16(
            false, qf[c], false, kf, (short)0, sc[ns], false, false);
      }

    float rmax[8];
    #pragma unroll
    for (int r = 0; r < 8; ++r) {
      const int qrow = q0 + hh * 8 + r;
      float v0 = sc[0][r] * 0.125f;
      float v1 = sc[1][r] * 0.125f;
      if (kb + l15 > qrow)      v0 = -__builtin_inff();
      if (kb + 16 + l15 > qrow) v1 = -__builtin_inff();
      sc[0][r] = v0; sc[1][r] = v1;
      float mx = fmaxf(v0, v1);
      #pragma unroll
      for (int m = 8; m >= 1; m >>= 1) mx = fmaxf(mx, __shfl_xor(mx, m, 32));
      rmax[r] = mx;
    }

    #pragma unroll
    for (int r = 0; r < 8; ++r) {
      const float mnew = fmaxf(mrow[r], rmax[r]);
      const float corr = __expf(mrow[r] - mnew);
      mrow[r] = mnew;
      const float p0 = __expf(sc[0][r] - mnew);
      const float p1 = __expf(sc[1][r] - mnew);
      sc[0][r] = p0; sc[1][r] = p1;
      float rs = p0 + p1;
      #pragma unroll
      for (int m = 8; m >= 1; m >>= 1) rs += __shfl_xor(rs, m, 32);
      lrow[r] = lrow[r] * corr + rs;
      #pragma unroll
      for (int t = 0; t < 4; ++t) o[t][r] *= corr;
    }

    // P: D-layout -> A-layout via wave-private LDS (per-wave DS is in-order)
    #pragma unroll
    for (int ns = 0; ns < 2; ++ns)
      #pragma unroll
      for (int r = 0; r < 8; ++r)
        sP[wave][hh * 8 + r][ns * 16 + l15] = (bf16)sc[ns][r];
    asm volatile("s_wait_dscnt 0x0" ::: "memory");
    v16bf pf;
    {
      const bf16* p = &sP[wave][l15][hh * 8];
      pf = concat8(*(const v8bf*)p, *(const v8bf*)(p + 16));
    }

    #pragma unroll
    for (int dt = 0; dt < 4; ++dt) {
      const bf16* p = &VT[vbase + (size_t)(dt * 16 + l15) * SLEN + kb + hh * 8];
      const v16bf vf = concat8(*(const v8bf*)p, *(const v8bf*)(p + 16));
      o[dt] = __builtin_amdgcn_wmma_f32_16x16x32_bf16(
          false, pf, false, vf, (short)0, o[dt], false, false);
    }
  }

  const int b = bh >> 4, h = bh & 15;
  #pragma unroll
  for (int r = 0; r < 8; ++r) {
    const float inv = 1.0f / lrow[r];
    const int srow = q0 + hh * 8 + r;
    #pragma unroll
    for (int dt = 0; dt < 4; ++dt)
      CTX[((size_t)(b * SLEN + srow)) * DM + h * HD + dt * 16 + l15] =
          (bf16)(o[dt][r] * inv);
  }
}

// ---------------------------------------------------------------------------
extern "C" void kernel_launch(void* const* d_in, const int* in_sizes, int n_in,
                              void* d_out, int out_size, void* d_ws, size_t ws_size,
                              hipStream_t stream) {
  (void)in_sizes; (void)n_in; (void)out_size; (void)ws_size;
  const float* x  = (const float*)d_in[0];
  const float* Wq = (const float*)d_in[1];
  const float* Wk = (const float*)d_in[2];
  const float* Wv = (const float*)d_in[3];
  const float* Wo = (const float*)d_in[4];
  const float* bo = (const float*)d_in[5];

  // workspace (bf16): xb 16MB | 4x Wt 8MB | Q,K,V^T,ctx 64MB  => 88 MB
  bf16* xb  = (bf16*)d_ws;
  bf16* wqt = xb  + (size_t)BSROWS * DM;
  bf16* wkt = wqt + (size_t)DM * DM;
  bf16* wvt = wkt + (size_t)DM * DM;
  bf16* wot = wvt + (size_t)DM * DM;
  bf16* q   = wot + (size_t)DM * DM;
  bf16* k   = q   + (size_t)BSROWS * DM;
  bf16* vt  = k   + (size_t)BSROWS * DM;
  bf16* ctx = vt  + (size_t)BSROWS * DM;

  const dim3 gConvX((BSROWS * DM) / (256 * 8));            // 4096
  const dim3 gConvW(DM / 32, DM / 32);                     // 32 x 32
  const dim3 bConvW(32, 8);
  const dim3 gGemm(BSROWS / 128, DM / 128);                // 64 x 8
  const dim3 gAttn(SLEN / 128, BATCH * NH);                // 16 x 64

  convert_x_kernel <<<gConvX, 256, 0, stream>>>(x, xb);
  convert_wT_kernel<<<gConvW, bConvW, 0, stream>>>(Wq, wqt);
  convert_wT_kernel<<<gConvW, bConvW, 0, stream>>>(Wk, wkt);
  convert_wT_kernel<<<gConvW, bConvW, 0, stream>>>(Wv, wvt);
  convert_wT_kernel<<<gConvW, bConvW, 0, stream>>>(Wo, wot);

  gemm_bf16_kernel<<<gGemm, 256, 0, stream>>>(xb, wqt, q,  nullptr, nullptr, 0);
  gemm_bf16_kernel<<<gGemm, 256, 0, stream>>>(xb, wkt, k,  nullptr, nullptr, 0);
  gemm_bf16_kernel<<<gGemm, 256, 0, stream>>>(xb, wvt, vt, nullptr, nullptr, 1);
  attn_kernel     <<<gAttn, 256, 0, stream>>>(q, k, vt, ctx);
  gemm_bf16_kernel<<<gGemm, 256, 0, stream>>>(ctx, wot, nullptr, (float*)d_out, bo, 2);
}
